// MultiHeadAttention_63161789055234
// MI455X (gfx1250) — compile-verified, matrix operand order checked by reading, not run on hardware
//
#include <hip/hip_runtime.h>

// ---------------------------------------------------------------------------
// Types for CDNA5 WMMA (wave32)
// ---------------------------------------------------------------------------
typedef __bf16 bf16;
typedef __attribute__((ext_vector_type(16))) __bf16 v16bf;
typedef __attribute__((ext_vector_type(8)))  __bf16 v8bf;
typedef __attribute__((ext_vector_type(8)))  float  v8f;

union VFrag { v16bf v; v8bf h[2]; };

static __device__ __forceinline__ v8f wmma_bf16(const VFrag& a, const VFrag& b, v8f c) {
  // D(16x16,f32) = A(16x32,bf16) * B(32x16,bf16) + C
  return __builtin_amdgcn_wmma_f32_16x16x32_bf16(
      /*neg_a=*/false, a.v, /*neg_b=*/false, b.v,
      /*c_mod=*/(short)0, c, /*reuse_a=*/false, /*reuse_b=*/false);
}

static __device__ __forceinline__ v8bf cvt8(const float4 f0, const float4 f1) {
  v8bf a;
  a[0] = (bf16)f0.x; a[1] = (bf16)f0.y; a[2] = (bf16)f0.z; a[3] = (bf16)f0.w;
  a[4] = (bf16)f1.x; a[5] = (bf16)f1.y; a[6] = (bf16)f1.z; a[7] = (bf16)f1.w;
  return a;
}

// --- async global -> LDS copy (16B per lane), tracked by ASYNCcnt ----------
static __device__ __forceinline__ void async_copy16(void* lds_dst, const void* gsrc) {
  unsigned loff = (unsigned)(size_t)(__attribute__((address_space(3))) char*)lds_dst;
  asm volatile("global_load_async_to_lds_b128 %0, %1, off"
               :: "v"(loff), "v"(gsrc) : "memory");
}
static __device__ __forceinline__ void wait_async0() {
  asm volatile("s_wait_asynccnt 0" ::: "memory");
}

// --- cross-lane xor within 16-lane groups via v_permlane16_b32 (no LDS) ----
static __device__ __forceinline__ float permx16(float x, unsigned s0, unsigned s1) {
  unsigned u = __float_as_uint(x);
  u = __builtin_amdgcn_permlane16(u, u, s0, s1, false, false);
  return __uint_as_float(u);
}
static __device__ __forceinline__ float redmax16(float x) {
  x = fmaxf(x, permx16(x, 0x67452301u, 0xEFCDAB89u));  // xor 1
  x = fmaxf(x, permx16(x, 0x45670123u, 0xCDEF89ABu));  // xor 3
  x = fmaxf(x, permx16(x, 0x32107654u, 0xBA98FEDCu));  // xor 4
  x = fmaxf(x, permx16(x, 0xFEDCBA98u, 0x76543210u));  // xor 8
  return x;
}
static __device__ __forceinline__ float redsum16(float x) {
  x += permx16(x, 0x67452301u, 0xEFCDAB89u);
  x += permx16(x, 0x45670123u, 0xCDEF89ABu);
  x += permx16(x, 0x32107654u, 0xBA98FEDCu);
  x += permx16(x, 0xFEDCBA98u, 0x76543210u);
  return x;
}

// ---------------------------------------------------------------------------
// Problem constants (bs=2, s=2048, d=1024, h=16, d_k=64)
// ---------------------------------------------------------------------------
constexpr int BS  = 2;
constexpr int SQ  = 2048;
constexpr int D   = 1024;
constexpr int H   = 16;
constexpr int DKH = 64;
constexpr int M_ROWS = BS * SQ;   // 4096

// GEMM tiling: 128x64 block tile, 8 waves each computing a 32x32 sub-tile
constexpr int BM = 128, BN = 64, BK = 32;
constexpr int LDT = 40;           // padded LDS row stride (elems) -> 80B

// Output modes for the projection GEMM
constexpr int OUT_F32       = 0;  // fp32 [row, col]
constexpr int OUT_HEADS     = 1;  // bf16 [b, h, s, d_k]
constexpr int OUT_HEADS_T   = 2;  // bf16 [b, h, d_k, s]   (for V: feeds PV directly)

// ---------------------------------------------------------------------------
// GEMM: Out = A @ W^T   (A: [M_ROWS, D], W: [D, D] row-major -> W row n = out col n)
// Double-buffered LDS pipeline; async copy for bf16 A; 4 WMMA per wave/k-step.
// ---------------------------------------------------------------------------
template<bool A_BF16, int OUTMODE>
__global__ __launch_bounds__(256)
void gemm_xwt_kernel(const void* __restrict__ Aptr, const float* __restrict__ W,
                     void* __restrict__ Out)
{
  __shared__ __align__(16) bf16 As[2][BM * LDT];   // 20480 B
  __shared__ __align__(16) bf16 Bs[2][BN * LDT];   // 10240 B

  const int mbase = blockIdx.x * BM;
  const int nbase = blockIdx.y * BN;
  const int t    = threadIdx.x;
  const int lane = t & 31, wave = t >> 5;
  const int wm = wave & 3, wn = wave >> 2;          // 4 x 2 waves of 32x32
  const int r  = lane & 15, hi = lane >> 4;
  const int e  = hi ? 8 : 0;

  // staging coordinates
  const int arow = t >> 1;            // 0..127
  const int acol = (t & 1) * 16;      // 0 or 16
  const int brow = t >> 2;            // 0..63
  const int bcol = (t & 3) * 8;       // 0,8,16,24

  v8f c00 = {}, c01 = {}, c10 = {}, c11 = {};
  v8bf ra0, ra1, rb0;                 // staged next tile (registers, fp32 paths)

  auto stage_loads = [&](int k0, int pp) {
    if (A_BF16) {
      // direct async DMA: global bf16 -> LDS, no registers
      const bf16* Ab = (const bf16*)Aptr + (size_t)(mbase + arow) * D + k0 + acol;
      async_copy16(&As[pp][arow * LDT + acol],     Ab);
      async_copy16(&As[pp][arow * LDT + acol + 8], Ab + 8);
    } else {
      const float* Af = (const float*)Aptr + (size_t)(mbase + arow) * D + k0 + acol;
      const float4 f0 = *(const float4*)Af;
      const float4 f1 = *(const float4*)(Af + 4);
      const float4 f2 = *(const float4*)(Af + 8);
      const float4 f3 = *(const float4*)(Af + 12);
      ra0 = cvt8(f0, f1);
      ra1 = cvt8(f2, f3);
    }
    const float* Wp = W + (size_t)(nbase + brow) * D + k0 + bcol;
    const float4 g0 = *(const float4*)Wp;
    const float4 g1 = *(const float4*)(Wp + 4);
    rb0 = cvt8(g0, g1);
    if (k0 + BK < D)
      __builtin_prefetch(Wp + BK, 0, 1);   // global_prefetch next W tile
  };
  auto stage_stores = [&](int pp) {
    if (!A_BF16) {
      *(v8bf*)&As[pp][arow * LDT + acol]     = ra0;
      *(v8bf*)&As[pp][arow * LDT + acol + 8] = ra1;
    }
    *(v8bf*)&Bs[pp][brow * LDT + bcol] = rb0;
  };

  stage_loads(0, 0);
  stage_stores(0);
  wait_async0();
  __syncthreads();

  int p = 0;
  for (int k0 = 0; k0 < D; k0 += BK) {
    const bool has_next = (k0 + BK < D);
    if (has_next) stage_loads(k0 + BK, p ^ 1);   // async + global loads in flight

    VFrag a0, a1, b0, b1;
    const int ao0 = (wm * 32 + r) * LDT;
    const int ao1 = (wm * 32 + 16 + r) * LDT;
    a0.h[0] = *(const v8bf*)&As[p][ao0 + e];
    a0.h[1] = *(const v8bf*)&As[p][ao0 + 16 + e];
    a1.h[0] = *(const v8bf*)&As[p][ao1 + e];
    a1.h[1] = *(const v8bf*)&As[p][ao1 + 16 + e];
    const int bo0 = (wn * 32 + r) * LDT;
    const int bo1 = (wn * 32 + 16 + r) * LDT;
    b0.h[0] = *(const v8bf*)&Bs[p][bo0 + e];
    b0.h[1] = *(const v8bf*)&Bs[p][bo0 + 16 + e];
    b1.h[0] = *(const v8bf*)&Bs[p][bo1 + e];
    b1.h[1] = *(const v8bf*)&Bs[p][bo1 + 16 + e];

    c00 = wmma_bf16(a0, b0, c00);
    c01 = wmma_bf16(a0, b1, c01);
    c10 = wmma_bf16(a1, b0, c10);
    c11 = wmma_bf16(a1, b1, c11);

    if (has_next) stage_stores(p ^ 1);
    wait_async0();
    __syncthreads();
    p ^= 1;
  }

  // --- store: D-tile layout = lane%16 -> col, vgpr i -> row (hi lanes rows 8..15)
  const int col = r;
  const int mr0 = hi * 8;
#pragma unroll
  for (int i = 0; i < 8; ++i) {
    const int grow0 = mbase + wm * 32 + mr0 + i;       // c00 / c01
    const int grow1 = grow0 + 16;                      // c10 / c11
    const int gc0   = nbase + wn * 32 + col;
    const int gc1   = gc0 + 16;
    if (OUTMODE == OUT_F32) {
      float* O = (float*)Out;
      O[(size_t)grow0 * D + gc0] = c00[i];
      O[(size_t)grow0 * D + gc1] = c01[i];
      O[(size_t)grow1 * D + gc0] = c10[i];
      O[(size_t)grow1 * D + gc1] = c11[i];
    } else {
      bf16* O = (bf16*)Out;
      const int h0 = gc0 >> 6, w0 = gc0 & 63;
      const int h1 = gc1 >> 6, w1 = gc1 & 63;
      const int b0i = grow0 >> 11, sr0 = grow0 & (SQ - 1);
      const int b1i = grow1 >> 11, sr1 = grow1 & (SQ - 1);
      if (OUTMODE == OUT_HEADS) {
        O[(((size_t)b0i * H + h0) * SQ + sr0) * DKH + w0] = (bf16)c00[i];
        O[(((size_t)b0i * H + h1) * SQ + sr0) * DKH + w1] = (bf16)c01[i];
        O[(((size_t)b1i * H + h0) * SQ + sr1) * DKH + w0] = (bf16)c10[i];
        O[(((size_t)b1i * H + h1) * SQ + sr1) * DKH + w1] = (bf16)c11[i];
      } else {  // OUT_HEADS_T : [b, h, d_k, s]
        O[(((size_t)b0i * H + h0) * DKH + w0) * SQ + sr0] = (bf16)c00[i];
        O[(((size_t)b0i * H + h1) * DKH + w1) * SQ + sr0] = (bf16)c01[i];
        O[(((size_t)b1i * H + h0) * DKH + w0) * SQ + sr1] = (bf16)c10[i];
        O[(((size_t)b1i * H + h1) * DKH + w1) * SQ + sr1] = (bf16)c11[i];
      }
    }
  }
}

// ---------------------------------------------------------------------------
// Flash attention: per (b,h), online-softmax over key tiles of 32.
// Block = 256 threads = 8 waves, each wave owns 16 q rows (block: 128 q rows).
// K and V^T tiles staged by async DMA (double-buffered); permlane16 reductions.
// V is pre-transposed by its projection GEMM: Vh layout is [b, h, d_k, s].
// ---------------------------------------------------------------------------
constexpr int KT  = 32;   // key tile
constexpr int LDK = 72;   // Ks row stride (64 dk + pad)
constexpr int LDV = 40;   // Vt row stride (32 keys + pad)
constexpr int LDP = 40;   // Ps row stride (32 keys + pad)

__global__ __launch_bounds__(256)
void attention_kernel(const bf16* __restrict__ Qh, const bf16* __restrict__ Kh,
                      const bf16* __restrict__ Vt_g, const int* __restrict__ mask,
                      bf16* __restrict__ attnOut)
{
  __shared__ __align__(16) bf16 Ks[2][KT * LDK];      // [key][dk]       9216 B
  __shared__ __align__(16) bf16 Vt[2][DKH * LDV];     // [dk][key]      10240 B
  __shared__ __align__(16) bf16 Ps[8 * 16 * LDP];     // per-wave P     10240 B

  const int bi   = blockIdx.x;
  const int qblk = bi & 15;            // SQ/128 = 16 q-blocks
  const int bh   = bi >> 4;            // 0..31
  const int b    = bh >> 4;
  const int head = bh & 15;
  const int qbase = qblk * 128;

  const int t    = threadIdx.x;
  const int lane = t & 31, wave = t >> 5;
  const int r    = lane & 15, hi = lane >> 4;
  const int e    = hi ? 8 : 0;

  // --- Q A-fragments, loaded once from global (two 32-wide k-steps over dk=64)
  VFrag aq0, aq1;
  {
    const bf16* Qrow = Qh + ((size_t)bh * SQ + qbase + wave * 16 + r) * DKH;
    aq0.h[0] = *(const v8bf*)&Qrow[e];
    aq0.h[1] = *(const v8bf*)&Qrow[16 + e];
    aq1.h[0] = *(const v8bf*)&Qrow[32 + e];
    aq1.h[1] = *(const v8bf*)&Qrow[48 + e];
  }

  float mrow[8], lrow[8];
  v8f o0 = {}, o1 = {}, o2 = {}, o3 = {};
#pragma unroll
  for (int i = 0; i < 8; ++i) { mrow[i] = -3.0e38f; lrow[i] = 0.0f; }

  // staging coordinates (16B async copy per lane per tensor)
  const int krow = t >> 3;         // 0..31  (key row)
  const int kcol = (t & 7) * 8;    // 0..56  (dk)
  const int vrow = t >> 2;         // 0..63  (dk row of V^T)
  const int vcol = (t & 3) * 8;    // 0,8,16,24 (keys)

  auto issue_kv = [&](int kb, int pp) {
    const bf16* gk = Kh   + ((size_t)bh * SQ  + kb + krow) * DKH + kcol;
    async_copy16(&Ks[pp][krow * LDK + kcol], gk);
    const bf16* gv = Vt_g + ((size_t)bh * DKH + vrow) * SQ + kb + vcol;
    async_copy16(&Vt[pp][vrow * LDV + vcol], gv);
  };

  issue_kv(0, 0);
  wait_async0();
  __syncthreads();

  int p = 0;
  for (int kb = 0; kb < SQ; kb += KT) {
    const bool has_next = (kb + KT < SQ);
    if (has_next) issue_kv(kb + KT, p ^ 1);   // async DMA in flight during WMMA

    // per-lane mask bias for this tile's two 16-key columns
    const float bias0 = (mask[b * SQ + kb + r]      == 0) ? -1.0e9f : 0.0f;
    const float bias1 = (mask[b * SQ + kb + 16 + r] == 0) ? -1.0e9f : 0.0f;

    // --- S = Q K^T : two 16-col tiles, contraction dk=64 (2 k-steps) ---
    v8f s0 = {}, s1 = {};
    {
      VFrag bk;
      const int row0 = r * LDK;
      bk.h[0] = *(const v8bf*)&Ks[p][row0 + e];
      bk.h[1] = *(const v8bf*)&Ks[p][row0 + 16 + e];
      s0 = wmma_bf16(aq0, bk, s0);
      bk.h[0] = *(const v8bf*)&Ks[p][row0 + 32 + e];
      bk.h[1] = *(const v8bf*)&Ks[p][row0 + 48 + e];
      s0 = wmma_bf16(aq1, bk, s0);
      const int row1 = (16 + r) * LDK;
      bk.h[0] = *(const v8bf*)&Ks[p][row1 + e];
      bk.h[1] = *(const v8bf*)&Ks[p][row1 + 16 + e];
      s1 = wmma_bf16(aq0, bk, s1);
      bk.h[0] = *(const v8bf*)&Ks[p][row1 + 32 + e];
      bk.h[1] = *(const v8bf*)&Ks[p][row1 + 48 + e];
      s1 = wmma_bf16(aq1, bk, s1);
    }

    // --- online softmax (rows in vgpr index; cols across 16-lane groups) ---
    bf16* Psw = &Ps[wave * 16 * LDP];
#pragma unroll
    for (int i = 0; i < 8; ++i) {
      float si0 = s0[i] * 0.125f + bias0;     // 1/sqrt(64) = 0.125
      float si1 = s1[i] * 0.125f + bias1;
      float mx = redmax16(fmaxf(si0, si1));
      const float mn = fmaxf(mrow[i], mx);
      const float al = __expf(mrow[i] - mn);
      const float p0 = __expf(si0 - mn);
      const float p1 = __expf(si1 - mn);
      const float rs = redsum16(p0 + p1);
      lrow[i] = lrow[i] * al + rs;
      mrow[i] = mn;
      o0[i] *= al; o1[i] *= al; o2[i] *= al; o3[i] *= al;
      // spill P (D-tile layout) to wave-private LDS for A-fragment re-layout
      Psw[(hi * 8 + i) * LDP + r]      = (bf16)p0;
      Psw[(hi * 8 + i) * LDP + 16 + r] = (bf16)p1;
    }

    // --- O += P @ V : A = P(16x32 keys), B = V^T rows (dk x keys) ---
    VFrag ap;
    ap.h[0] = *(const v8bf*)&Psw[r * LDP + e];
    ap.h[1] = *(const v8bf*)&Psw[r * LDP + 16 + e];
    {
      VFrag bv;
#define PV_STEP(T, ACC)                                             \
      { const int row = ((T) * 16 + r) * LDV;                       \
        bv.h[0] = *(const v8bf*)&Vt[p][row + e];                    \
        bv.h[1] = *(const v8bf*)&Vt[p][row + 16 + e];               \
        ACC = wmma_bf16(ap, bv, ACC); }
      PV_STEP(0, o0) PV_STEP(1, o1) PV_STEP(2, o2) PV_STEP(3, o3)
#undef PV_STEP
    }

    wait_async0();
    __syncthreads();
    p ^= 1;
  }

  // --- normalize and store bf16 into concat layout [b, s, h*64 + dk] ---
#pragma unroll
  for (int i = 0; i < 8; ++i) {
    const float inv = 1.0f / lrow[i];
    const int row = qbase + wave * 16 + hi * 8 + i;
    bf16* orow = attnOut + ((size_t)b * SQ + row) * D + head * DKH;
    orow[r]      = (bf16)(o0[i] * inv);
    orow[16 + r] = (bf16)(o1[i] * inv);
    orow[32 + r] = (bf16)(o2[i] * inv);
    orow[48 + r] = (bf16)(o3[i] * inv);
  }
}

// ---------------------------------------------------------------------------
// Launch: q/k/v projections -> flash attention -> output projection
// ---------------------------------------------------------------------------
extern "C" void kernel_launch(void* const* d_in, const int* in_sizes, int n_in,
                              void* d_out, int out_size, void* d_ws, size_t ws_size,
                              hipStream_t stream) {
  const float* k    = (const float*)d_in[0];
  const float* v    = (const float*)d_in[1];
  const float* q    = (const float*)d_in[2];
  const int*   mask = (const int*)d_in[3];
  const float* Wq   = (const float*)d_in[4];
  const float* Wk   = (const float*)d_in[5];
  const float* Wv   = (const float*)d_in[6];
  const float* Wo   = (const float*)d_in[7];

  const size_t HEAD_ELEMS = (size_t)BS * H * SQ * DKH;   // 4,194,304 (8 MB bf16)
  bf16* Qh    = (bf16*)d_ws;
  bf16* Kh    = Qh + HEAD_ELEMS;
  bf16* Vh    = Kh + HEAD_ELEMS;   // stored transposed: [b, h, d_k, s]
  bf16* attnB = Vh + HEAD_ELEMS;

  dim3 ggrid(M_ROWS / BM, D / BN);   // (32, 16)

  gemm_xwt_kernel<false, OUT_HEADS  ><<<ggrid, 256, 0, stream>>>(q, Wq, Qh);
  gemm_xwt_kernel<false, OUT_HEADS  ><<<ggrid, 256, 0, stream>>>(k, Wk, Kh);
  gemm_xwt_kernel<false, OUT_HEADS_T><<<ggrid, 256, 0, stream>>>(v, Wv, Vh);

  attention_kernel<<<BS * H * (SQ / 128), 256, 0, stream>>>(Qh, Kh, Vh, mask, attnB);

  gemm_xwt_kernel<true, OUT_F32><<<ggrid, 256, 0, stream>>>(attnB, Wo, d_out);

  (void)in_sizes; (void)n_in; (void)out_size; (void)ws_size;
}
